// AttentionBlock_60739427500827
// MI455X (gfx1250) — matrix operand order repaired and checked by reading, not compiled, not run on hardware
//
#include <hip/hip_runtime.h>
#include <hip/hip_bf16.h>

typedef __attribute__((ext_vector_type(16))) __bf16       v16bf;
typedef __attribute__((ext_vector_type(8)))  float        v8f;
typedef __attribute__((ext_vector_type(4)))  float        f32x4;
typedef __attribute__((ext_vector_type(4)))  unsigned int u32x4;
typedef __attribute__((ext_vector_type(2)))  unsigned int u32x2;

#define LSTR 40   // ushorts per LDS row: 32 payload + 8 pad (keeps 16B alignment)

__device__ __forceinline__ unsigned short f2bf(float f) {
  unsigned int u = __builtin_bit_cast(unsigned int, f);
  u += 0x7FFFu + ((u >> 16) & 1u);              // round-to-nearest-even
  return (unsigned short)(u >> 16);
}

// LDS byte address for async-to-LDS: generic pointers to LDS keep the LDS
// offset in addr[31:0] (ISA 10.2 aperture mapping), so truncation is exact.
__device__ __forceinline__ unsigned ldsOff(const void* p) {
  return (unsigned)(unsigned long long)p;
}

// GLOBAL_LOAD_ASYNC_TO_LDS_B128: per-lane 16B copy global->LDS, ASYNCcnt.
__device__ __forceinline__ void asyncLd128(unsigned ldsByte, const void* g) {
  asm volatile("global_load_async_to_lds_b128 %0, %1, off"
               :: "v"(ldsByte), "v"((unsigned long long)g) : "memory");
}
__device__ __forceinline__ void waitAsync0() {
  asm volatile("s_wait_asynccnt 0x0" ::: "memory");
}

// A fragment: 16x32 bf16. lane m = lane%16, hf = lane/16.
__device__ __forceinline__ v16bf fragA(const unsigned short* L, int row, int hf) {
  union { u32x4 q[2]; v16bf v; } u;
  const unsigned short* p = L + row * LSTR;
  u.q[0] = *(const u32x4*)(p + hf * 8);
  u.q[1] = *(const u32x4*)(p + 16 + hf * 8);
  return u.v;
}

// B fragment: 32x16 bf16. lane n = lane%16, element e<8 -> k = hf*8 + e, e>=8 -> k = 16 + hf*8 + (e-8).
__device__ __forceinline__ v16bf fragB(const unsigned short* L, int row, int hf) {
  union { u32x4 q[2]; v16bf v; } u;
  const unsigned short* p = L + row * LSTR;
  u.q[0] = *(const u32x4*)(p + hf * 8);
  u.q[1] = *(const u32x4*)(p + 16 + hf * 8);
  return u.v;
}

// One K-slab of MMA work for a wave: 4x2 tiles of 16x16, Kc=32.
__device__ __forceinline__ void mma_step(const unsigned short* Abuf,
                                         const unsigned short* Bbuf,
                                         int wm, int wn, int lane16, int hf,
                                         v8f acc[4][2]) {
  v16bf a[4], b[2];
#pragma unroll
  for (int mt = 0; mt < 4; ++mt) a[mt] = fragA(Abuf, wm * 64 + mt * 16 + lane16, hf);
#pragma unroll
  for (int nt = 0; nt < 2; ++nt) b[nt] = fragB(Bbuf, wn * 32 + nt * 16 + lane16, hf);
#pragma unroll
  for (int mt = 0; mt < 4; ++mt)
#pragma unroll
    for (int nt = 0; nt < 2; ++nt)
      acc[mt][nt] = __builtin_amdgcn_wmma_f32_16x16x32_bf16(
          false, a[mt], false, b[nt], (short)0, acc[mt][nt], false, false);
}

// ---------------------------------------------------------------------------
// Kernel 1: QKV projection. out[s,d] = sum_c X[s,c]*W[d,c] + b[d], bf16 out.
// z==0/1 (Q,K): row-major [s][d].  z==2 (V): TRANSPOSED out Vt[b][d][s] so
// kernel 4's B operand needs no LDS transpose (and the store is packed b128).
// grid (64, 8, 3), block 256. Register-staged double-buffered pipeline
// (fp32 -> bf16 conversion happens in-flight).
// ---------------------------------------------------------------------------
__device__ __forceinline__ void k1_load(const float* __restrict__ X,
                                        const float* __restrict__ W,
                                        int m0, int n0, int k0, int tid,
                                        f32x4 ar[4], f32x4 br[4]) {
#pragma unroll
  for (int i = 0; i < 4; ++i) {
    int f = tid + i * 256;
    int row = f >> 3, c4 = f & 7;
    ar[i] = *(const f32x4*)(X + (size_t)(m0 + row) * 1024 + k0 + c4 * 4);
    br[i] = *(const f32x4*)(W + (size_t)(n0 + row) * 1024 + k0 + c4 * 4);
  }
}

__device__ __forceinline__ void k1_store(unsigned short* Abuf, unsigned short* Bbuf,
                                         int tid, const f32x4 ar[4], const f32x4 br[4]) {
#pragma unroll
  for (int i = 0; i < 4; ++i) {
    int f = tid + i * 256;
    int row = f >> 3, c4 = f & 7;
    u32x2 pa = { ((unsigned)f2bf(ar[i].y) << 16) | f2bf(ar[i].x),
                 ((unsigned)f2bf(ar[i].w) << 16) | f2bf(ar[i].z) };
    u32x2 pb = { ((unsigned)f2bf(br[i].y) << 16) | f2bf(br[i].x),
                 ((unsigned)f2bf(br[i].w) << 16) | f2bf(br[i].z) };
    *(u32x2*)(Abuf + row * LSTR + c4 * 4) = pa;
    *(u32x2*)(Bbuf + row * LSTR + c4 * 4) = pb;
  }
}

__global__ __launch_bounds__(256)
void qkv_proj_kernel(const float* __restrict__ q, const float* __restrict__ k,
                     const float* __restrict__ v,
                     const float* __restrict__ Wq, const float* __restrict__ bq,
                     const float* __restrict__ Wk, const float* __restrict__ bk,
                     const float* __restrict__ Wv, const float* __restrict__ bv,
                     unsigned short* __restrict__ qkv_out)
{
  const int z = blockIdx.z;
  const float* X    = (z == 0) ? q  : (z == 1) ? k  : v;
  const float* W    = (z == 0) ? Wq : (z == 1) ? Wk : Wv;
  const float* bias = (z == 0) ? bq : (z == 1) ? bk : bv;
  unsigned short* out = qkv_out + (size_t)z * (8192u * 1024u);

  __shared__ __align__(16) unsigned short Al[2][128 * LSTR];
  __shared__ __align__(16) unsigned short Bl[2][128 * LSTR];

  const int tid = threadIdx.x;
  const int lane = tid & 31, wid = tid >> 5;
  const int wm = wid >> 2, wn = wid & 3;
  const int lane16 = lane & 15, hf = lane >> 4;
  const int m0 = blockIdx.x * 128;
  const int n0 = blockIdx.y * 128;

  const v8f vzero = {0.f, 0.f, 0.f, 0.f, 0.f, 0.f, 0.f, 0.f};
  v8f acc[4][2];
#pragma unroll
  for (int i = 0; i < 4; ++i)
#pragma unroll
    for (int j = 0; j < 2; ++j) acc[i][j] = vzero;

  f32x4 ar[4], br[4];
  k1_load(X, W, m0, n0, 0, tid, ar, br);
  k1_store(Al[0], Bl[0], tid, ar, br);
  __syncthreads();

  for (int kt = 0; kt < 32; ++kt) {
    const int cur = kt & 1;
    if (kt + 1 < 32) k1_load(X, W, m0, n0, (kt + 1) * 32, tid, ar, br);
    if (kt + 2 < 32) {
#pragma unroll
      for (int i = 0; i < 4; ++i) {
        int f = tid + i * 256;
        int row = f >> 3, c4 = f & 7;
        __builtin_prefetch(X + (size_t)(m0 + row) * 1024 + (kt + 2) * 32 + c4 * 4, 0, 1);
        __builtin_prefetch(W + (size_t)(n0 + row) * 1024 + (kt + 2) * 32 + c4 * 4, 0, 1);
      }
    }
    mma_step(Al[cur], Bl[cur], wm, wn, lane16, hf, acc);
    if (kt + 1 < 32) {
      k1_store(Al[1 - cur], Bl[1 - cur], tid, ar, br);
      __syncthreads();
    }
  }

  if (z != 2) {
    // Q,K: row-major bf16 [s][d]
#pragma unroll
    for (int mt = 0; mt < 4; ++mt)
#pragma unroll
      for (int nt = 0; nt < 2; ++nt) {
        int col = n0 + wn * 32 + nt * 16 + lane16;
        float bvv = bias[col];
#pragma unroll
        for (int r = 0; r < 8; ++r) {
          int row = m0 + wm * 64 + mt * 16 + hf * 8 + r;
          out[(size_t)row * 1024 + col] = f2bf(acc[mt][nt][r] + bvv);
        }
      }
  } else {
    // V: transposed Vt[b][d][s]; lane's 8 rows are s-consecutive -> one b128
#pragma unroll
    for (int mt = 0; mt < 4; ++mt)
#pragma unroll
      for (int nt = 0; nt < 2; ++nt) {
        int col = n0 + wn * 32 + nt * 16 + lane16;
        float bvv = bias[col];
        int row0 = m0 + wm * 64 + mt * 16 + hf * 8;     // 8-aligned
        unsigned pk[4];
#pragma unroll
        for (int j = 0; j < 4; ++j)
          pk[j] = ((unsigned)f2bf(acc[mt][nt][2 * j + 1] + bvv) << 16) |
                  f2bf(acc[mt][nt][2 * j] + bvv);
        u32x4 val = {pk[0], pk[1], pk[2], pk[3]};
        size_t base = ((size_t)(row0 >> 11) * 1024 + col) * 2048 + (row0 & 2047);
        *(u32x4*)(out + base) = val;
      }
  }
}

// ---------------------------------------------------------------------------
// Kernel 2: scores = (Q . K^T) / sqrt(C), fp32 out. grid (16,16,4), block 256.
// Async-to-LDS double-buffered pipeline (bf16 operands copied without VGPRs).
// ---------------------------------------------------------------------------
__global__ __launch_bounds__(256)
void scores_kernel(const unsigned short* __restrict__ Qb,
                   const unsigned short* __restrict__ Kb,
                   float* __restrict__ Sc)
{
  const int batch = blockIdx.z;
  const unsigned short* Qp = Qb + (size_t)batch * 2048 * 1024;
  const unsigned short* Kp = Kb + (size_t)batch * 2048 * 1024;
  float* S = Sc + (size_t)batch * 2048 * 2048;

  __shared__ __align__(16) unsigned short Al[2][128 * LSTR];
  __shared__ __align__(16) unsigned short Bl[2][128 * LSTR];

  const int tid = threadIdx.x;
  const int lane = tid & 31, wid = tid >> 5;
  const int wm = wid >> 2, wn = wid & 3;
  const int lane16 = lane & 15, hf = lane >> 4;
  const int m0 = blockIdx.x * 128;
  const int n0 = blockIdx.y * 128;

  const int rowL = tid >> 2, c8 = tid & 3;            // 4 x b128 per 32-wide row
  const unsigned short* gA  = Qp + (size_t)(m0 + rowL) * 1024 + c8 * 8;
  const unsigned short* gA2 = Qp + (size_t)(m0 + 64 + rowL) * 1024 + c8 * 8;
  const unsigned short* gB  = Kp + (size_t)(n0 + rowL) * 1024 + c8 * 8;
  const unsigned short* gB2 = Kp + (size_t)(n0 + 64 + rowL) * 1024 + c8 * 8;

  unsigned aOff[2], a2Off[2], bOff[2], b2Off[2];
#pragma unroll
  for (int i = 0; i < 2; ++i) {
    aOff[i]  = ldsOff(Al[i] + rowL * LSTR + c8 * 8);
    a2Off[i] = ldsOff(Al[i] + (64 + rowL) * LSTR + c8 * 8);
    bOff[i]  = ldsOff(Bl[i] + rowL * LSTR + c8 * 8);
    b2Off[i] = ldsOff(Bl[i] + (64 + rowL) * LSTR + c8 * 8);
  }

  const v8f vzero = {0.f, 0.f, 0.f, 0.f, 0.f, 0.f, 0.f, 0.f};
  v8f acc[4][2];
#pragma unroll
  for (int i = 0; i < 4; ++i)
#pragma unroll
    for (int j = 0; j < 2; ++j) acc[i][j] = vzero;

  asyncLd128(aOff[0], gA);   asyncLd128(a2Off[0], gA2);
  asyncLd128(bOff[0], gB);   asyncLd128(b2Off[0], gB2);
  waitAsync0();
  __syncthreads();

  for (int kt = 0; kt < 32; ++kt) {
    const int cur = kt & 1;
    if (kt + 1 < 32) {
      const int k0 = (kt + 1) * 32;
      asyncLd128(aOff[1 - cur],  gA + k0);
      asyncLd128(a2Off[1 - cur], gA2 + k0);
      asyncLd128(bOff[1 - cur],  gB + k0);
      asyncLd128(b2Off[1 - cur], gB2 + k0);
    }
    mma_step(Al[cur], Bl[cur], wm, wn, lane16, hf, acc);
    if (kt + 1 < 32) {
      waitAsync0();
      __syncthreads();
    }
  }

  const float scale = 0.03125f;  // 1/sqrt(1024)
#pragma unroll
  for (int mt = 0; mt < 4; ++mt)
#pragma unroll
    for (int nt = 0; nt < 2; ++nt) {
      int col = n0 + wn * 32 + nt * 16 + lane16;
#pragma unroll
      for (int r = 0; r < 8; ++r) {
        int row = m0 + wm * 64 + mt * 16 + hf * 8 + r;
        S[(size_t)row * 2048 + col] = acc[mt][nt][r] * scale;
      }
    }
}

// ---------------------------------------------------------------------------
// Kernel 3: row softmax over 2048 fp32 -> bf16 P. One wave per row.
// grid (1024), block 256 (8 rows/block).
// ---------------------------------------------------------------------------
__global__ __launch_bounds__(256)
void softmax_kernel(const float* __restrict__ Sc, unsigned short* __restrict__ P)
{
  const int lane = threadIdx.x & 31;
  const int row = blockIdx.x * 8 + (threadIdx.x >> 5);   // 0..8191
  const float* src = Sc + (size_t)row * 2048;
  unsigned short* dst = P + (size_t)row * 2048;

  float v[64];
  float mx = -__builtin_inff();
#pragma unroll
  for (int i = 0; i < 64; ++i) {
    v[i] = src[i * 32 + lane];
    mx = fmaxf(mx, v[i]);
  }
#pragma unroll
  for (int off = 16; off > 0; off >>= 1)
    mx = fmaxf(mx, __shfl_xor(mx, off, 32));

  float s = 0.f;
#pragma unroll
  for (int i = 0; i < 64; ++i) {
    v[i] = __expf(v[i] - mx);
    s += v[i];
  }
#pragma unroll
  for (int off = 16; off > 0; off >>= 1)
    s += __shfl_xor(s, off, 32);

  const float inv = 1.0f / s;
#pragma unroll
  for (int i = 0; i < 64; ++i)
    dst[i * 32 + lane] = f2bf(v[i] * inv);
}

// ---------------------------------------------------------------------------
// Kernel 4: out = P . V, fp32 out. grid (16,8,4), block 256.
// V is pre-transposed (Vt[b][d][s]) so both operand fills are plain row
// copies -> async-to-LDS double-buffered pipeline.
// ---------------------------------------------------------------------------
__global__ __launch_bounds__(256)
void av_kernel(const unsigned short* __restrict__ P,
               const unsigned short* __restrict__ Vt,
               float* __restrict__ out)
{
  const int batch = blockIdx.z;
  const unsigned short* Pp = P  + (size_t)batch * 2048 * 2048;
  const unsigned short* Vp = Vt + (size_t)batch * 1024 * 2048;
  float* O = out + (size_t)batch * 2048 * 1024;

  __shared__ __align__(16) unsigned short Al[2][128 * LSTR];
  __shared__ __align__(16) unsigned short Bl[2][128 * LSTR];

  const int tid = threadIdx.x;
  const int lane = tid & 31, wid = tid >> 5;
  const int wm = wid >> 2, wn = wid & 3;
  const int lane16 = lane & 15, hf = lane >> 4;
  const int m0 = blockIdx.x * 128;   // q rows
  const int n0 = blockIdx.y * 128;   // d cols

  const int rowL = tid >> 2, c8 = tid & 3;
  const unsigned short* gA  = Pp + (size_t)(m0 + rowL) * 2048 + c8 * 8;
  const unsigned short* gA2 = Pp + (size_t)(m0 + 64 + rowL) * 2048 + c8 * 8;
  const unsigned short* gB  = Vp + (size_t)(n0 + rowL) * 2048 + c8 * 8;       // Vt rows (d)
  const unsigned short* gB2 = Vp + (size_t)(n0 + 64 + rowL) * 2048 + c8 * 8;

  unsigned aOff[2], a2Off[2], bOff[2], b2Off[2];
#pragma unroll
  for (int i = 0; i < 2; ++i) {
    aOff[i]  = ldsOff(Al[i] + rowL * LSTR + c8 * 8);
    a2Off[i] = ldsOff(Al[i] + (64 + rowL) * LSTR + c8 * 8);
    bOff[i]  = ldsOff(Bl[i] + rowL * LSTR + c8 * 8);
    b2Off[i] = ldsOff(Bl[i] + (64 + rowL) * LSTR + c8 * 8);
  }

  const v8f vzero = {0.f, 0.f, 0.f, 0.f, 0.f, 0.f, 0.f, 0.f};
  v8f acc[4][2];
#pragma unroll
  for (int i = 0; i < 4; ++i)
#pragma unroll
    for (int j = 0; j < 2; ++j) acc[i][j] = vzero;

  asyncLd128(aOff[0], gA);   asyncLd128(a2Off[0], gA2);
  asyncLd128(bOff[0], gB);   asyncLd128(b2Off[0], gB2);
  waitAsync0();
  __syncthreads();

  for (int kt = 0; kt < 64; ++kt) {
    const int cur = kt & 1;
    if (kt + 1 < 64) {
      const int k0 = (kt + 1) * 32;
      asyncLd128(aOff[1 - cur],  gA + k0);
      asyncLd128(a2Off[1 - cur], gA2 + k0);
      asyncLd128(bOff[1 - cur],  gB + k0);
      asyncLd128(b2Off[1 - cur], gB2 + k0);
    }
    mma_step(Al[cur], Bl[cur], wm, wn, lane16, hf, acc);
    if (kt + 1 < 64) {
      waitAsync0();
      __syncthreads();
    }
  }

#pragma unroll
  for (int mt = 0; mt < 4; ++mt)
#pragma unroll
    for (int nt = 0; nt < 2; ++nt) {
      int col = n0 + wn * 32 + nt * 16 + lane16;
#pragma unroll
      for (int r = 0; r < 8; ++r) {
        int row = m0 + wm * 64 + mt * 16 + hf * 8 + r;
        O[(size_t)row * 1024 + col] = acc[mt][nt][r];
      }
    }
}

// ---------------------------------------------------------------------------
extern "C" void kernel_launch(void* const* d_in, const int* in_sizes, int n_in,
                              void* d_out, int out_size, void* d_ws, size_t ws_size,
                              hipStream_t stream) {
  (void)in_sizes; (void)n_in; (void)out_size; (void)ws_size;
  const float* query = (const float*)d_in[0];
  const float* key_t = (const float*)d_in[1];
  const float* value = (const float*)d_in[2];
  const float* Wq    = (const float*)d_in[3];
  const float* bq    = (const float*)d_in[4];
  const float* Wk    = (const float*)d_in[5];
  const float* bk    = (const float*)d_in[6];
  const float* Wv    = (const float*)d_in[7];
  const float* bv    = (const float*)d_in[8];

  const size_t QKV_E = (size_t)8192 * 1024;          // elements per bf16 tensor
  unsigned short* Qb = (unsigned short*)d_ws;
  unsigned short* Kb = Qb + QKV_E;
  unsigned short* Vb = Kb + QKV_E;                   // holds Vt[b][d][s]
  float*          Sc = (float*)(Vb + QKV_E);         // 64 MB fp32 scores
  unsigned short* Pm = (unsigned short*)(Sc + (size_t)4 * 2048 * 2048);

  dim3 blk(256);
  qkv_proj_kernel<<<dim3(64, 8, 3),  blk, 0, stream>>>(query, key_t, value,
                                                       Wq, bq, Wk, bk, Wv, bv, Qb);
  scores_kernel  <<<dim3(16, 16, 4), blk, 0, stream>>>(Qb, Kb, Sc);
  softmax_kernel <<<dim3(1024),      blk, 0, stream>>>(Sc, Pm);
  av_kernel      <<<dim3(16, 8, 4),  blk, 0, stream>>>(Pm, Vb, (float*)d_out);
}
